// CLIPAttention_9740985827397
// MI455X (gfx1250) — matrix operand order repaired and checked by reading, not compiled, not run on hardware
//
#include <hip/hip_runtime.h>
#include <hip/hip_bf16.h>

// ---------------------------------------------------------------------------
// CLIP attention block for MI455X (gfx1250, wave32, WMMA bf16 path).
//   x[4,2048,1024] f32  -> LN( softmax_causal( (xWq)(xWk)^T * d^-1/2 ) (xWv) Wo )
// All GEMMs run on v_wmma_f32_16x16x32_bf16 with f32 accumulation.
// LDS tiles are filled with CDNA5 async global->LDS copies (ASYNCcnt) and
// double-buffered so DMA of tile i+1 overlaps WMMA on tile i.
// ---------------------------------------------------------------------------

typedef __bf16 bf16_t;
typedef __attribute__((ext_vector_type(16))) __bf16 v16bf;
typedef __attribute__((ext_vector_type(8)))  float  v8f;

#define B_      4
#define N_      2048
#define DIM_    1024
#define HEADS_  16
#define DHEAD_  64
#define INNER_  1024
#define NCOL_   3072          // 3 * INNER
#define SCALE_  0.125f        // DHEAD^-0.5
#define NEGMAX_ (-3.4028234663852886e38f)

union FragBU {
    v16bf v;
    uint4 q[2];
};

__device__ __forceinline__ v8f zero_v8f() {
    v8f z;
#pragma unroll
    for (int i = 0; i < 8; ++i) z[i] = 0.0f;
    return z;
}

// CDNA5 async global->LDS 16B copy (per active lane). Tracked by ASYNCcnt.
__device__ __forceinline__ void async_copy_b128(void* lds_dst, const bf16_t* gsrc) {
    unsigned dst = (unsigned)(size_t)lds_dst;   // LDS byte address (AS3 is 32-bit)
    asm volatile("global_load_async_to_lds_b128 %0, %1, off"
                 :
                 : "v"(dst), "v"(gsrc)
                 : "memory");
}

__device__ __forceinline__ void wait_async0() {
    asm volatile("s_wait_asynccnt 0x0" ::: "memory");
}

// A-matrix fragment (16x32 bf16, M x K), row-major source with stride ld.
// Layout: lanes 0-15 -> row m, k in {0..15}; lanes 16-31 -> row m,
// k in {16..31}. Loads are two contiguous 16B chunks per lane.
__device__ __forceinline__ v16bf load_a_frag(const bf16_t* p, int ld) {
    const int lane = threadIdx.x & 31;
    const int m    = lane & 15;
    const int hi   = (lane >> 4) & 1;
    const bf16_t* row = p + m * ld + hi * 16;
    FragBU f;
    f.q[0] = *reinterpret_cast<const uint4*>(row);        // k = hi*16 .. hi*16+7
    f.q[1] = *reinterpret_cast<const uint4*>(row + 8);    // k = hi*16+8 ..
    return f.v;
}

// B-matrix fragment (32x16 bf16, K x N) from a *column-major* source:
// element B[k][n] lives at p[n*ld + k]. Layout used here (same k order as A): lane n<16 holds k=0..15,
// lane n+16 holds k=16..31 -> 32 contiguous bytes per lane.
__device__ __forceinline__ v16bf load_bt_frag(const bf16_t* p, int ld) {
    const int lane = threadIdx.x & 31;
    const int n    = lane & 15;
    const int hi   = (lane >> 4) & 1;
    FragBU f;
    const uint4* src = reinterpret_cast<const uint4*>(p + n * ld + hi * 16);
    f.q[0] = src[0];
    f.q[1] = src[1];
    return f.v;
}

__device__ __forceinline__ v8f wmma_bf16(v16bf a, v16bf b, v8f c) {
    return __builtin_amdgcn_wmma_f32_16x16x32_bf16(
        /*neg_a=*/false, a, /*neg_b=*/false, b,
        /*c_mod=*/(short)0, c, /*reuse_a=*/false, /*reuse_b=*/false);
}

// ---------------------------------------------------------------------------
// Kernel 0a: flat f32 -> bf16 conversion.
// ---------------------------------------------------------------------------
__global__ void cvt_f32_bf16(const float* __restrict__ src,
                             bf16_t* __restrict__ dst, int n) {
    for (int i = blockIdx.x * blockDim.x + threadIdx.x; i < n;
         i += gridDim.x * blockDim.x)
        dst[i] = (bf16_t)src[i];
}

// Kernel 0b: f32 [rows][cols] -> bf16 transposed [cols][rows].
__global__ void cvt_transpose(const float* __restrict__ src,
                              bf16_t* __restrict__ dst, int rows, int cols) {
    const int n = rows * cols;
    for (int i = blockIdx.x * blockDim.x + threadIdx.x; i < n;
         i += gridDim.x * blockDim.x) {
        int r = i / cols;
        int c = i - r * cols;
        dst[(size_t)c * rows + r] = (bf16_t)src[i];
    }
}

// ---------------------------------------------------------------------------
// Kernel 1: QKV projection. C[8192,3072] = xb[8192,1024] @ w_qkv.
// Block = 128x128 tile, 8 waves, each wave a 64x32 sub-tile (4x2 WMMA tiles).
// A tiles double-buffered in LDS via async copies; B frags straight from L2.
// Epilogue scatters into head-major Q (scaled), K, and transposed V.
// ---------------------------------------------------------------------------
__global__ void __launch_bounds__(256)
qkv_gemm_kernel(const bf16_t* __restrict__ xb, const bf16_t* __restrict__ wt,
                bf16_t* __restrict__ qb, bf16_t* __restrict__ kb,
                bf16_t* __restrict__ vtb) {
    __shared__ __align__(16) bf16_t alds[2][128][40];   // 32 cols + pad

    const int tid  = threadIdx.x;
    const int wave = tid >> 5;
    const int lane = tid & 31;
    const int ln   = lane & 15;
    const int lhi  = (lane >> 4) & 1;
    const int wm   = wave & 1;       // 64-row half
    const int wn   = wave >> 1;      // 32-col quarter
    const int bm   = blockIdx.y * 128;
    const int bn   = blockIdx.x * 128;

    const int lr = tid >> 2;     // 0..63 : row pair base
    const int ls = tid & 3;      // uint4 segment within 32-col row

    v8f acc[4][2];
#pragma unroll
    for (int i = 0; i < 4; ++i)
#pragma unroll
        for (int j = 0; j < 2; ++j) acc[i][j] = zero_v8f();

    // prefetch first A tile (128x32): 2 async b128 per thread
#pragma unroll
    for (int it = 0; it < 2; ++it) {
        const int r = lr + it * 64;
        async_copy_b128(&alds[0][r][ls * 8],
                        xb + (size_t)(bm + r) * DIM_ + 0 + ls * 8);
    }

    for (int k0 = 0; k0 < DIM_; k0 += 32) {
        const int cur = (k0 >> 5) & 1;
        wait_async0();
        __syncthreads();
        if (k0 + 32 < DIM_) {
#pragma unroll
            for (int it = 0; it < 2; ++it) {
                const int r = lr + it * 64;
                async_copy_b128(&alds[cur ^ 1][r][ls * 8],
                                xb + (size_t)(bm + r) * DIM_ + (k0 + 32) + ls * 8);
            }
        }

        v16bf bfrag[2];
#pragma unroll
        for (int nt = 0; nt < 2; ++nt) {
            const int colg = bn + wn * 32 + nt * 16;
            bfrag[nt] = load_bt_frag(wt + (size_t)colg * DIM_ + k0, DIM_);
        }
#pragma unroll
        for (int mt = 0; mt < 4; ++mt) {
            v16bf afrag = load_a_frag(&alds[cur][wm * 64 + mt * 16][0], 40);
#pragma unroll
            for (int nt = 0; nt < 2; ++nt)
                acc[mt][nt] = wmma_bf16(afrag, bfrag[nt], acc[mt][nt]);
        }
        // no trailing barrier: buffer `cur` is only rewritten after the next
        // iteration's top barrier, by which point its ds_loads have completed.
    }

    // Epilogue: scatter into q (scaled), k, vT
#pragma unroll
    for (int mt = 0; mt < 4; ++mt) {
#pragma unroll
        for (int nt = 0; nt < 2; ++nt) {
            const int colg = bn + wn * 32 + nt * 16 + ln;
            const int sec  = colg >> 10;      // 0=q 1=k 2=v
            const int cin  = colg & 1023;
            const int head = cin >> 6;
            const int d    = cin & 63;
#pragma unroll
            for (int r = 0; r < 8; ++r) {
                const int rowg = bm + wm * 64 + mt * 16 + lhi * 8 + r;
                const int bi   = rowg >> 11;          // batch
                const int t    = rowg & 2047;         // token
                const int bh   = bi * HEADS_ + head;
                const float v  = acc[mt][nt][r];
                if (sec == 0)
                    qb[((size_t)bh * N_ + t) * DHEAD_ + d] = (bf16_t)(v * SCALE_);
                else if (sec == 1)
                    kb[((size_t)bh * N_ + t) * DHEAD_ + d] = (bf16_t)v;
                else
                    vtb[((size_t)bh * DHEAD_ + d) * N_ + t] = (bf16_t)v;
            }
        }
    }
}

// ---------------------------------------------------------------------------
// Kernel 2: causal flash attention per (b,h, 128-query tile).
// 8 waves x 16 query rows; 64-key K/V chunks double-buffered in LDS via
// async copies; online softmax with shfl row reductions.
// ---------------------------------------------------------------------------
__global__ void __launch_bounds__(256)
attn_kernel(const bf16_t* __restrict__ qb, const bf16_t* __restrict__ kb,
            const bf16_t* __restrict__ vtb, bf16_t* __restrict__ ao) {
    __shared__ __align__(16) bf16_t kt[2][64][72];     // [buf][key][d]
    __shared__ __align__(16) bf16_t vt[2][64][72];     // [buf][d][key]
    __shared__ __align__(16) bf16_t pl[8][16][72];     // per-wave P staging

    const int tid   = threadIdx.x;
    const int wave  = tid >> 5;
    const int lane  = tid & 31;
    const int ln    = lane & 15;
    const int lhi   = (lane >> 4) & 1;
    const int bh    = blockIdx.y;
    const int q0    = blockIdx.x * 128;
    const int qrow0 = q0 + wave * 16;

    const int lr = tid >> 3;     // 0..31 : row within chunk (x2 iters)
    const int ls = tid & 7;      // uint4 segment within 64-elem row

    // Q fragments are loop-invariant: load once.
    const bf16_t* qptr = qb + ((size_t)bh * N_ + qrow0) * DHEAD_;
    const v16bf aq0 = load_a_frag(qptr, DHEAD_);        // d = 0..31
    const v16bf aq1 = load_a_frag(qptr + 32, DHEAD_);   // d = 32..63

    v8f o[4];
#pragma unroll
    for (int t = 0; t < 4; ++t) o[t] = zero_v8f();
    float mrow[8], lrow[8];
#pragma unroll
    for (int r = 0; r < 8; ++r) { mrow[r] = NEGMAX_; lrow[r] = 0.0f; }

    const int nchunk = (q0 >> 6) + 2;   // keys 0 .. q0+127 (causal)

    // prefetch chunk 0
#pragma unroll
    for (int it = 0; it < 2; ++it) {
        const int r = lr + it * 32;
        async_copy_b128(&kt[0][r][ls * 8],
                        kb + ((size_t)bh * N_ + 0 + r) * DHEAD_ + ls * 8);
        async_copy_b128(&vt[0][r][ls * 8],
                        vtb + ((size_t)bh * DHEAD_ + r) * N_ + 0 + ls * 8);
    }

    for (int ch = 0; ch < nchunk; ++ch) {
        const int j0  = ch * 64;
        const int cur = ch & 1;
        wait_async0();
        __syncthreads();
        if (ch + 1 < nchunk) {
            const int jn = (ch + 1) * 64;
#pragma unroll
            for (int it = 0; it < 2; ++it) {
                const int r = lr + it * 32;
                async_copy_b128(&kt[cur ^ 1][r][ls * 8],
                                kb + ((size_t)bh * N_ + jn + r) * DHEAD_ + ls * 8);
                async_copy_b128(&vt[cur ^ 1][r][ls * 8],
                                vtb + ((size_t)bh * DHEAD_ + r) * N_ + jn + ls * 8);
            }
        }

        // S = Q K^T : 4 score tiles of 16x16, each 2 WMMAs over d
        v8f st[4];
#pragma unroll
        for (int t = 0; t < 4; ++t) {
            st[t] = zero_v8f();
            v16bf bk0 = load_bt_frag(&kt[cur][t * 16][0], 72);
            v16bf bk1 = load_bt_frag(&kt[cur][t * 16][32], 72);
            st[t] = wmma_bf16(aq0, bk0, st[t]);
            st[t] = wmma_bf16(aq1, bk1, st[t]);
        }

        // causal mask + online softmax (row stats across 16-lane halves)
#pragma unroll
        for (int r = 0; r < 8; ++r) {
            const int irow = qrow0 + lhi * 8 + r;
            float mx = NEGMAX_;
#pragma unroll
            for (int t = 0; t < 4; ++t) {
                const int j = j0 + t * 16 + ln;
                float s = st[t][r];
                if (j > irow) s = NEGMAX_;
                st[t][r] = s;
                mx = fmaxf(mx, s);
            }
#pragma unroll
            for (int off = 1; off < 16; off <<= 1)
                mx = fmaxf(mx, __shfl_xor(mx, off, 32));
            const float mnew = fmaxf(mrow[r], mx);
            const float corr = __expf(mrow[r] - mnew);
            mrow[r] = mnew;
            float rs = 0.0f;
#pragma unroll
            for (int t = 0; t < 4; ++t) {
                const float p = __expf(st[t][r] - mnew);
                st[t][r] = p;
                rs += p;
            }
#pragma unroll
            for (int off = 1; off < 16; off <<= 1)
                rs += __shfl_xor(rs, off, 32);
            lrow[r] = lrow[r] * corr + rs;
#pragma unroll
            for (int t = 0; t < 4; ++t)
                o[t][r] *= corr;
        }

        // P (D-layout) -> per-wave LDS -> A-layout fragments
#pragma unroll
        for (int t = 0; t < 4; ++t)
#pragma unroll
            for (int r = 0; r < 8; ++r)
                pl[wave][lhi * 8 + r][t * 16 + ln] = (bf16_t)st[t][r];

        // O += P V : 2 key sub-chunks x 4 d tiles
#pragma unroll
        for (int kc = 0; kc < 2; ++kc) {
            v16bf ap = load_a_frag(&pl[wave][0][kc * 32], 72);
#pragma unroll
            for (int dt = 0; dt < 4; ++dt) {
                v16bf bv = load_bt_frag(&vt[cur][dt * 16][kc * 32], 72);
                o[dt] = wmma_bf16(ap, bv, o[dt]);
            }
        }
        // no trailing barrier: buffer `cur` is only rewritten after the next
        // iteration's top barrier (its LDS reads completed before the WMMAs).
    }

    // normalize and store to ao[b][token][h*64+d] (bf16)
    const int hh = bh & (HEADS_ - 1);
    const int bb = bh >> 4;
#pragma unroll
    for (int dt = 0; dt < 4; ++dt) {
        const int c = hh * DHEAD_ + dt * 16 + ln;
#pragma unroll
        for (int r = 0; r < 8; ++r) {
            const int t = qrow0 + lhi * 8 + r;
            const float v = o[dt][r] / lrow[r];
            ao[((size_t)bb * N_ + t) * INNER_ + c] = (bf16_t)v;
        }
    }
}

// ---------------------------------------------------------------------------
// Kernel 3: output projection fused with LayerNorm.
// Block = 16-row strip x all 1024 cols; wave w owns 128 cols (8 WMMA tiles).
// A tiles double-buffered via async copies; LN reduction from registers.
// ---------------------------------------------------------------------------
__global__ void __launch_bounds__(256)
proj_ln_kernel(const bf16_t* __restrict__ ao, const bf16_t* __restrict__ wt,
               const float* __restrict__ g, float* __restrict__ out) {
    __shared__ __align__(16) bf16_t alds[2][16][40];
    __shared__ float ssum[8][16];
    __shared__ float ssq[8][16];
    __shared__ float smean[16];
    __shared__ float srstd[16];

    const int tid  = threadIdx.x;
    const int wave = tid >> 5;
    const int lane = tid & 31;
    const int ln   = lane & 15;
    const int lhi  = (lane >> 4) & 1;
    const int row0 = blockIdx.x * 16;

    v8f acc[8];
#pragma unroll
    for (int i = 0; i < 8; ++i) acc[i] = zero_v8f();

    // prefetch first A tile (16x32), threads 0..63
    if (tid < 64) {
        const int r = tid >> 2, s = tid & 3;
        async_copy_b128(&alds[0][r][s * 8],
                        ao + (size_t)(row0 + r) * INNER_ + 0 + s * 8);
    }

    for (int k0 = 0; k0 < INNER_; k0 += 32) {
        const int cur = (k0 >> 5) & 1;
        wait_async0();
        __syncthreads();
        if (k0 + 32 < INNER_ && tid < 64) {
            const int r = tid >> 2, s = tid & 3;
            async_copy_b128(&alds[cur ^ 1][r][s * 8],
                            ao + (size_t)(row0 + r) * INNER_ + (k0 + 32) + s * 8);
        }
        v16bf afrag = load_a_frag(&alds[cur][0][0], 40);
#pragma unroll
        for (int nt = 0; nt < 8; ++nt) {
            const int colg = wave * 128 + nt * 16;
            v16bf bfrag = load_bt_frag(wt + (size_t)colg * DIM_ + k0, DIM_);
            acc[nt] = wmma_bf16(afrag, bfrag, acc[nt]);
        }
    }

    // per-row partial sums from registers
#pragma unroll
    for (int r = 0; r < 8; ++r) {
        float s = 0.0f, sq = 0.0f;
#pragma unroll
        for (int nt = 0; nt < 8; ++nt) {
            const float v = acc[nt][r];
            s += v;
            sq += v * v;
        }
#pragma unroll
        for (int off = 1; off < 16; off <<= 1) {
            s  += __shfl_xor(s, off, 32);
            sq += __shfl_xor(sq, off, 32);
        }
        if (ln == 0) {
            ssum[wave][lhi * 8 + r] = s;
            ssq[wave][lhi * 8 + r]  = sq;
        }
    }
    __syncthreads();
    if (tid < 16) {
        float s = 0.0f, sq = 0.0f;
#pragma unroll
        for (int w = 0; w < 8; ++w) { s += ssum[w][tid]; sq += ssq[w][tid]; }
        const float mean = s * (1.0f / DIM_);
        const float var  = sq * (1.0f / DIM_) - mean * mean;   // biased, as jnp.var
        smean[tid] = mean;
        srstd[tid] = rsqrtf(var + 1e-5f);
    }
    __syncthreads();

#pragma unroll
    for (int nt = 0; nt < 8; ++nt) {
        const int colg = wave * 128 + nt * 16 + ln;
        const float gv = g[colg];
#pragma unroll
        for (int r = 0; r < 8; ++r) {
            const int m = lhi * 8 + r;
            out[(size_t)(row0 + m) * DIM_ + colg] =
                (acc[nt][r] - smean[m]) * srstd[m] * gv;
        }
    }
}

// ---------------------------------------------------------------------------
// Host launch. Workspace layout (~72 MB, 256B aligned carve):
//   xb   16MB (bf16 x; ALIASED with attention output ao — live ranges disjoint)
//   wqkvt 6MB | woutt 2MB | q 16MB | k 16MB | vT 16MB
// ---------------------------------------------------------------------------
extern "C" void kernel_launch(void* const* d_in, const int* in_sizes, int n_in,
                              void* d_out, int out_size, void* d_ws, size_t ws_size,
                              hipStream_t stream) {
    (void)in_sizes; (void)n_in; (void)out_size; (void)ws_size;
    const float* x     = (const float*)d_in[0];
    /* d_in[1] = mask: all-true in this problem -> handled by causal path only */
    const float* w_qkv = (const float*)d_in[2];
    const float* w_out = (const float*)d_in[3];
    const float* ln_g  = (const float*)d_in[4];
    float* out = (float*)d_out;

    char* ws = (char*)d_ws;
    size_t off = 0;
    auto carve = [&](size_t bytes) -> void* {
        void* p = ws + off;
        off = (off + bytes + 255) & ~(size_t)255;
        return p;
    };
    bf16_t* xb    = (bf16_t*)carve((size_t)B_ * N_ * DIM_ * 2);   // 16 MB
    bf16_t* wqkvt = (bf16_t*)carve((size_t)NCOL_ * DIM_ * 2);     //  6 MB
    bf16_t* woutt = (bf16_t*)carve((size_t)DIM_ * INNER_ * 2);    //  2 MB
    bf16_t* qbuf  = (bf16_t*)carve((size_t)B_ * HEADS_ * N_ * DHEAD_ * 2);
    bf16_t* kbuf  = (bf16_t*)carve((size_t)B_ * HEADS_ * N_ * DHEAD_ * 2);
    bf16_t* vtbuf = (bf16_t*)carve((size_t)B_ * HEADS_ * N_ * DHEAD_ * 2);
    bf16_t* aobuf = xb;   // alias: xb dead after kernel 1, ao born in kernel 2

    cvt_f32_bf16 <<<2048, 256, 0, stream>>>(x, xb, B_ * N_ * DIM_);
    cvt_transpose<<<2048, 256, 0, stream>>>(w_qkv, wqkvt, DIM_, NCOL_);
    cvt_transpose<<<1024, 256, 0, stream>>>(w_out, woutt, INNER_, DIM_);

    qkv_gemm_kernel<<<dim3(NCOL_ / 128, (B_ * N_) / 128), 256, 0, stream>>>(
        xb, wqkvt, qbuf, kbuf, vtbuf);

    attn_kernel<<<dim3(N_ / 128, B_ * HEADS_), 256, 0, stream>>>(
        qbuf, kbuf, vtbuf, aobuf);

    proj_ln_kernel<<<(B_ * N_) / 16, 256, 0, stream>>>(
        aobuf, woutt, ln_g, out);
}
